// TransformerBlock_20366734917908
// MI455X (gfx1250) — matrix operand order repaired and checked by reading, not compiled, not run on hardware
//
#include <hip/hip_runtime.h>
#include <math.h>

// ---------------------------------------------------------------------------
// Types
// ---------------------------------------------------------------------------
typedef unsigned short u16;                                   // bf16 bit storage
typedef __attribute__((ext_vector_type(16))) __bf16 v16bf;    // WMMA A/B operand (8 VGPRs)
typedef __attribute__((ext_vector_type(8)))  __bf16 bf16x8;   // 16-byte chunk
typedef __attribute__((ext_vector_type(8)))  float  v8f;      // WMMA C/D operand

__device__ __forceinline__ u16 f2b(float f) {                 // fp32 -> bf16 (RNE)
    union { float f; unsigned u; } x; x.f = f;
    unsigned r = (x.u + 0x7FFFu + ((x.u >> 16) & 1u)) >> 16;
    return (u16)r;
}

// CDNA5 async global->LDS DMA (ASYNCcnt-tracked, no VGPR round trip).
// LDS operand is the wave-relative LDS byte offset = low 32 bits of the flat
// address of a __shared__ pointer (LDS aperture occupies addr[63:32]).
__device__ __forceinline__ void async_ld_b128(const u16* gaddr, u16* lds) {
    asm volatile("global_load_async_to_lds_b128 %0, %1, off"
                 :: "v"((unsigned)(unsigned long long)lds), "v"(gaddr)
                 : "memory");
}

// ---------------------------------------------------------------------------
// Problem constants
// ---------------------------------------------------------------------------
#define BATCH 4
#define SEQ   2048
#define DMODEL 1024
#define DFF   4096
#define MTOT  (BATCH * SEQ)   // 8192

// GEMM tiling
#define BM 128
#define BN 128
#define BK 32
#define LDSK (BK + 8)         // 40 halves = 80B row stride, keeps 16B alignment

// Epilogue modes
#define EPI_F32_BIAS   0
#define EPI_B16_BIAS   1
#define EPI_B16_RELU   2
#define EPI_F32_SCALE  3
#define EPI_B16_PLAIN  4

// ---------------------------------------------------------------------------
// NT bf16 GEMM with WMMA:  C[m,n] = sum_k A[m,k] * B[n,k]   (+ epilogue)
// A: (M x K) bf16 row-major, B: (N x K) bf16 row-major. Batched via blockIdx.z.
// M, N multiples of 128; K multiple of 32.
// Double-buffered LDS fed by GLOBAL_LOAD_ASYNC_TO_LDS_B128.
// ---------------------------------------------------------------------------
template <int MODE>
__global__ __launch_bounds__(256) void gemm_nt_wmma(
    const u16* __restrict__ A, long sA,
    const u16* __restrict__ Bm, long sB,
    float* __restrict__ Cf, u16* __restrict__ Cb, long sC,
    const float* __restrict__ bias,
    int N, int K, float scale)
{
    __shared__ alignas(16) u16 lA[2][BM][LDSK];
    __shared__ alignas(16) u16 lB[2][BN][LDSK];

    const int z = blockIdx.z;
    A  += (long)z * sA;
    Bm += (long)z * sB;

    const int m0 = blockIdx.y * BM;
    const int n0 = blockIdx.x * BN;
    const int t    = threadIdx.x;
    const int wave = t >> 5;
    const int lane = t & 31;
    const int wr = wave >> 1;   // wave row 0..3 -> M offset wr*32
    const int wc = wave & 1;    // wave col 0..1 -> N offset wc*64
    const int r  = lane & 15;   // lane row/col within 16
    const int g  = lane >> 4;   // lane half-group

    v8f acc[2][4];
    #pragma unroll
    for (int i = 0; i < 2; ++i)
        #pragma unroll
        for (int j = 0; j < 4; ++j)
            acc[i][j] = (v8f){};

    // each thread stages 16 halves (2 x b128) per tile of each matrix
    const int lr   = t >> 1;          // 0..127
    const int lcol = (t & 1) * 16;    // 0 or 16
    const u16* gA = A  + (long)(m0 + lr) * K + lcol;
    const u16* gB = Bm + (long)(n0 + lr) * K + lcol;

    // prologue: async-stage tile 0 into buffer 0
    async_ld_b128(gA,     &lA[0][lr][lcol]);
    async_ld_b128(gA + 8, &lA[0][lr][lcol + 8]);
    async_ld_b128(gB,     &lB[0][lr][lcol]);
    async_ld_b128(gB + 8, &lB[0][lr][lcol + 8]);

    const int nk = K / BK;
    for (int kt = 0; kt < nk; ++kt) {
        const int cur = kt & 1;
        if (kt + 1 < nk) {
            // overlap: DMA tile kt+1 into the other buffer while computing kt.
            // Buffer (kt+1)&1 was last read in iter kt-1; the trailing barrier
            // of that iteration makes the overwrite safe.
            const int nxt = cur ^ 1;
            const u16* ga = gA + (long)(kt + 1) * BK;
            const u16* gb = gB + (long)(kt + 1) * BK;
            async_ld_b128(ga,     &lA[nxt][lr][lcol]);
            async_ld_b128(ga + 8, &lA[nxt][lr][lcol + 8]);
            async_ld_b128(gb,     &lB[nxt][lr][lcol]);
            async_ld_b128(gb + 8, &lB[nxt][lr][lcol + 8]);
            // async completions are in-order: <=4 outstanding => tile kt landed
            asm volatile("s_wait_asynccnt 0x4" ::: "memory");
        } else {
            asm volatile("s_wait_asynccnt 0x0" ::: "memory");
        }
        __syncthreads();

        // A fragment: lane holds row (lane%16); K chunks at 8g and 16+8g
        v16bf af[2];
        #pragma unroll
        for (int i = 0; i < 2; ++i) {
            const u16* p = &lA[cur][wr * 32 + i * 16 + r][0];
            bf16x8 lo = *(const bf16x8*)(p + g * 8);
            bf16x8 hi = *(const bf16x8*)(p + 16 + g * 8);
            af[i] = __builtin_shufflevector(lo, hi, 0,1,2,3,4,5,6,7,8,9,10,11,12,13,14,15);
        }
        // B fragment: lane holds col (lane%16); K chunks at 8g and 16+8g
        v16bf bfr[4];
        #pragma unroll
        for (int j = 0; j < 4; ++j) {
            const u16* p = &lB[cur][wc * 64 + j * 16 + r][0];
            bf16x8 lo = *(const bf16x8*)(p + g * 8);
            bf16x8 hi = *(const bf16x8*)(p + 16 + g * 8);
            bfr[j] = __builtin_shufflevector(lo, hi, 0,1,2,3,4,5,6,7,8,9,10,11,12,13,14,15);
        }

        #pragma unroll
        for (int i = 0; i < 2; ++i)
            #pragma unroll
            for (int j = 0; j < 4; ++j)
                acc[i][j] = __builtin_amdgcn_wmma_f32_16x16x32_bf16(
                    false, af[i], false, bfr[j], (short)0, acc[i][j], false, false);

        // all readers of buffer `cur` must pass before iter kt+1 DMAs into it
        __syncthreads();
    }

    // Epilogue: D tile layout — lanes 0-15: N=lane, VGPR p = row p;
    //                           lanes 16-31: N=lane-16, VGPR p = row p+8
    #pragma unroll
    for (int i = 0; i < 2; ++i) {
        #pragma unroll
        for (int j = 0; j < 4; ++j) {
            const int col = n0 + wc * 64 + j * 16 + r;
            float bv = 0.0f;
            if (MODE == EPI_F32_BIAS || MODE == EPI_B16_BIAS || MODE == EPI_B16_RELU)
                bv = bias[col];
            #pragma unroll
            for (int p = 0; p < 8; ++p) {
                const int row = m0 + wr * 32 + i * 16 + p + 8 * g;
                float v = acc[i][j][p];
                if (MODE == EPI_F32_SCALE) v *= scale;
                else                       v += bv;
                if (MODE == EPI_B16_RELU)  v = fmaxf(v, 0.0f);
                const long idx = (long)z * sC + (long)row * N + col;
                if (MODE == EPI_F32_BIAS || MODE == EPI_F32_SCALE) Cf[idx] = v;
                else                                               Cb[idx] = f2b(v);
            }
        }
    }
}

// ---------------------------------------------------------------------------
// fp32 -> bf16 elementwise
// ---------------------------------------------------------------------------
__global__ __launch_bounds__(256) void cvt_f32_to_bf16(const float* __restrict__ in,
                                                       u16* __restrict__ out, long n)
{
    long i = (long)blockIdx.x * 256 + threadIdx.x;
    if (i < n) out[i] = f2b(in[i]);
}

// fp32 W(K x N) -> bf16 W^T(N x K)
__global__ __launch_bounds__(256) void cvt_w_transpose(const float* __restrict__ W,
                                                       u16* __restrict__ WT, int K, int N)
{
    long i = (long)blockIdx.x * 256 + threadIdx.x;
    if (i < (long)K * N) {
        int k = (int)(i / N), n = (int)(i % N);
        WT[(long)n * K + k] = f2b(W[i]);
    }
}

// batched bf16 transpose: in (z, R, C) -> out (z, C, R)
__global__ __launch_bounds__(256) void transpose_bf16(const u16* __restrict__ in,
                                                      u16* __restrict__ out, int R, int C)
{
    __shared__ u16 tile[32][33];
    const long base = (long)blockIdx.z * R * C;
    const int c0 = blockIdx.x * 32, r0 = blockIdx.y * 32;
    const int tx = threadIdx.x & 31, ty = threadIdx.x >> 5;
    #pragma unroll
    for (int i = 0; i < 4; ++i) {
        int rr = ty + i * 8;
        tile[rr][tx] = in[base + (long)(r0 + rr) * C + c0 + tx];
    }
    __syncthreads();
    #pragma unroll
    for (int i = 0; i < 4; ++i) {
        int rr = ty + i * 8;
        out[base + (long)(c0 + rr) * R + r0 + tx] = tile[tx][rr];
    }
}

// ---------------------------------------------------------------------------
// Row softmax: fp32 scores (rows x 2048) -> bf16 probabilities
// ---------------------------------------------------------------------------
__global__ __launch_bounds__(256) void softmax_rows(const float* __restrict__ S,
                                                    u16* __restrict__ P)
{
    const int row = blockIdx.x;
    const int t = threadIdx.x;
    const float* s = S + (long)row * SEQ;
    u16* p = P + (long)row * SEQ;

    float v[8];
    float m = -INFINITY;
    #pragma unroll
    for (int i = 0; i < 8; ++i) { v[i] = s[t + i * 256]; m = fmaxf(m, v[i]); }

    __shared__ float red[256];
    red[t] = m; __syncthreads();
    for (int w = 128; w > 0; w >>= 1) {
        if (t < w) red[t] = fmaxf(red[t], red[t + w]);
        __syncthreads();
    }
    m = red[0]; __syncthreads();

    float sum = 0.0f;
    #pragma unroll
    for (int i = 0; i < 8; ++i) { v[i] = __expf(v[i] - m); sum += v[i]; }
    red[t] = sum; __syncthreads();
    for (int w = 128; w > 0; w >>= 1) {
        if (t < w) red[t] += red[t + w];
        __syncthreads();
    }
    const float inv = 1.0f / red[0];
    #pragma unroll
    for (int i = 0; i < 8; ++i) p[t + i * 256] = f2b(v[i] * inv);
}

// ---------------------------------------------------------------------------
// Fused residual-add + LayerNorm over rows of 1024; optional bf16 copy out
// ---------------------------------------------------------------------------
template <bool WRITE_B16>
__global__ __launch_bounds__(256) void add_layernorm(const float* __restrict__ X,
                                                     const float* __restrict__ Y,
                                                     const float* __restrict__ gamma,
                                                     const float* __restrict__ beta,
                                                     float* __restrict__ Of,
                                                     u16* __restrict__ Ob)
{
    const int row = blockIdx.x;
    const int t = threadIdx.x;
    const float* xr = X + (long)row * DMODEL;
    const float* yr = Y + (long)row * DMODEL;

    float v[4], s = 0.0f, ss = 0.0f;
    #pragma unroll
    for (int i = 0; i < 4; ++i) {
        int c = t + i * 256;
        v[i] = xr[c] + yr[c];
        s += v[i]; ss += v[i] * v[i];
    }
    __shared__ float r1[256], r2[256];
    r1[t] = s; r2[t] = ss; __syncthreads();
    for (int w = 128; w > 0; w >>= 1) {
        if (t < w) { r1[t] += r1[t + w]; r2[t] += r2[t + w]; }
        __syncthreads();
    }
    const float mean = r1[0] * (1.0f / DMODEL);
    const float var  = r2[0] * (1.0f / DMODEL) - mean * mean;
    const float rstd = rsqrtf(var + 1e-5f);
    #pragma unroll
    for (int i = 0; i < 4; ++i) {
        int c = t + i * 256;
        float o = (v[i] - mean) * rstd * gamma[c] + beta[c];
        Of[(long)row * DMODEL + c] = o;
        if (WRITE_B16) Ob[(long)row * DMODEL + c] = f2b(o);
    }
}

// ---------------------------------------------------------------------------
// Host orchestration
// ---------------------------------------------------------------------------
extern "C" void kernel_launch(void* const* d_in, const int* in_sizes, int n_in,
                              void* d_out, int out_size, void* d_ws, size_t ws_size,
                              hipStream_t stream)
{
    (void)in_sizes; (void)n_in; (void)out_size; (void)ws_size;

    const float* x   = (const float*)d_in[0];
    const float* wq  = (const float*)d_in[1];  const float* bq  = (const float*)d_in[2];
    const float* wk  = (const float*)d_in[3];  const float* bk  = (const float*)d_in[4];
    const float* wv  = (const float*)d_in[5];  const float* bv  = (const float*)d_in[6];
    const float* wo  = (const float*)d_in[7];  const float* bo  = (const float*)d_in[8];
    const float* w1  = (const float*)d_in[9];  const float* b1  = (const float*)d_in[10];
    const float* w2  = (const float*)d_in[11]; const float* b2  = (const float*)d_in[12];
    const float* g1  = (const float*)d_in[13]; const float* be1 = (const float*)d_in[14];
    const float* g2  = (const float*)d_in[15]; const float* be2 = (const float*)d_in[16];
    float* out = (float*)d_out;

    char* ws = (char*)d_ws;
    const size_t MB = 1024ull * 1024ull;
    // workspace map (with reuse; total 248 MB)
    u16*   Xb     = (u16*)(ws + 0 * MB);     // 16 MB  x bf16
    u16*   WqT    = (u16*)(ws + 16 * MB);    // 2 MB each
    u16*   WkT    = (u16*)(ws + 18 * MB);
    u16*   WvT    = (u16*)(ws + 20 * MB);
    u16*   WoT    = (u16*)(ws + 22 * MB);
    u16*   W1T    = (u16*)(ws + 24 * MB);    // 8 MB  (4096 x 1024)
    u16*   W2T    = (u16*)(ws + 32 * MB);    // 8 MB  (1024 x 4096)
    u16*   Qb     = (u16*)(ws + 40 * MB);    // 16 MB  (reused as x1 bf16)
    u16*   Kb     = (u16*)(ws + 56 * MB);    // 16 MB
    u16*   Vb     = (u16*)(ws + 72 * MB);    // 16 MB  (Vb+Vt reused as ff2 f32)
    u16*   Vt     = (u16*)(ws + 88 * MB);    // 16 MB
    float* scores = (float*)(ws + 104 * MB); // 64 MB  (reused as h1 bf16)
    u16*   attn   = (u16*)(ws + 168 * MB);   // 32 MB  (reused as x1 f32)
    u16*   ctx    = (u16*)(ws + 200 * MB);   // 16 MB
    float* attnO  = (float*)(ws + 216 * MB); // 32 MB
    // aliases (lifetimes are disjoint in the schedule below)
    u16*   h1   = (u16*)scores;              // 64 MB post-ReLU hidden, bf16
    float* x1   = (float*)attn;              // 32 MB
    u16*   x1b  = Qb;                        // 16 MB
    float* ff2  = (float*)Vb;                // 32 MB

    const long nXD = (long)MTOT * DMODEL;

    // --- 1. downconvert activations & weights (weights transposed to N x K) ---
    cvt_f32_to_bf16<<<dim3((unsigned)((nXD + 255) / 256)), 256, 0, stream>>>(x, Xb, nXD);
    const long nDD = (long)DMODEL * DMODEL, nDF = (long)DMODEL * DFF;
    cvt_w_transpose<<<dim3((unsigned)((nDD + 255) / 256)), 256, 0, stream>>>(wq, WqT, DMODEL, DMODEL);
    cvt_w_transpose<<<dim3((unsigned)((nDD + 255) / 256)), 256, 0, stream>>>(wk, WkT, DMODEL, DMODEL);
    cvt_w_transpose<<<dim3((unsigned)((nDD + 255) / 256)), 256, 0, stream>>>(wv, WvT, DMODEL, DMODEL);
    cvt_w_transpose<<<dim3((unsigned)((nDD + 255) / 256)), 256, 0, stream>>>(wo, WoT, DMODEL, DMODEL);
    cvt_w_transpose<<<dim3((unsigned)((nDF + 255) / 256)), 256, 0, stream>>>(w1, W1T, DMODEL, DFF);
    cvt_w_transpose<<<dim3((unsigned)((nDF + 255) / 256)), 256, 0, stream>>>(w2, W2T, DFF, DMODEL);

    // --- 2. Q/K/V projections: (8192x1024) x (1024x1024)^T, bias, -> bf16 ---
    dim3 gProj(DMODEL / BN, MTOT / BM, 1);
    gemm_nt_wmma<EPI_B16_BIAS><<<gProj, 256, 0, stream>>>(Xb, 0, WqT, 0, nullptr, Qb, 0, bq, DMODEL, DMODEL, 1.0f);
    gemm_nt_wmma<EPI_B16_BIAS><<<gProj, 256, 0, stream>>>(Xb, 0, WkT, 0, nullptr, Kb, 0, bk, DMODEL, DMODEL, 1.0f);
    gemm_nt_wmma<EPI_B16_BIAS><<<gProj, 256, 0, stream>>>(Xb, 0, WvT, 0, nullptr, Vb, 0, bv, DMODEL, DMODEL, 1.0f);

    // V^T per batch for the attn.V NT gemm
    transpose_bf16<<<dim3(DMODEL / 32, SEQ / 32, BATCH), 256, 0, stream>>>(Vb, Vt, SEQ, DMODEL);

    // --- 3. scores = Q K^T / sqrt(d)  (batched, f32) ---
    dim3 gScore(SEQ / BN, SEQ / BM, BATCH);
    gemm_nt_wmma<EPI_F32_SCALE><<<gScore, 256, 0, stream>>>(
        Qb, (long)SEQ * DMODEL, Kb, (long)SEQ * DMODEL,
        scores, nullptr, (long)SEQ * SEQ, nullptr, SEQ, DMODEL, 0.03125f);

    // --- 4. softmax rows -> bf16 probs ---
    softmax_rows<<<dim3(BATCH * SEQ), 256, 0, stream>>>(scores, attn);

    // --- 5. ctx = attn V  (batched NT against V^T) -> bf16 ---
    dim3 gCtx(DMODEL / BN, SEQ / BM, BATCH);
    gemm_nt_wmma<EPI_B16_PLAIN><<<gCtx, 256, 0, stream>>>(
        attn, (long)SEQ * SEQ, Vt, (long)DMODEL * SEQ,
        nullptr, ctx, (long)SEQ * DMODEL, nullptr, DMODEL, SEQ, 1.0f);

    // --- 6. attn_out = ctx Wo + bo  -> f32 ---
    gemm_nt_wmma<EPI_F32_BIAS><<<gProj, 256, 0, stream>>>(ctx, 0, WoT, 0, attnO, nullptr, 0, bo, DMODEL, DMODEL, 1.0f);

    // --- 7. x1 = LN(x + attn_out)  (f32 + bf16 copy for FFN input) ---
    add_layernorm<true><<<dim3(MTOT), 256, 0, stream>>>(x, attnO, g1, be1, x1, x1b);

    // --- 8. h1 = relu(x1 W1 + b1) -> bf16 ---
    dim3 gFF1(DFF / BN, MTOT / BM, 1);
    gemm_nt_wmma<EPI_B16_RELU><<<gFF1, 256, 0, stream>>>(x1b, 0, W1T, 0, nullptr, h1, 0, b1, DFF, DMODEL, 1.0f);

    // --- 9. ff2 = h1 W2 + b2 -> f32 ---
    gemm_nt_wmma<EPI_F32_BIAS><<<gProj, 256, 0, stream>>>(h1, 0, W2T, 0, ff2, nullptr, 0, b2, DMODEL, DFF, 1.0f);

    // --- 10. out = LN(x1 + ff2) ---
    add_layernorm<false><<<dim3(MTOT), 256, 0, stream>>>(x1, ff2, g2, be2, out, nullptr);
}
